// CrossAttention2d_21174188769755
// MI455X (gfx1250) — matrix operand order repaired and checked by reading, not compiled, not run on hardware
//
#include <hip/hip_runtime.h>

// ---------------------------------------------------------------------------
// CrossAttention2d for MI455X (gfx1250): f16 WMMA pipeline + async LDS staging
//   B=8, D_MODEL=512, D_COND=768, N_HEADS=8, dh=64, L=4096, L_COND=256
// ---------------------------------------------------------------------------

typedef __attribute__((ext_vector_type(16))) _Float16     v16h;
typedef __attribute__((ext_vector_type(8)))  float        v8f;
typedef __attribute__((ext_vector_type(4)))  unsigned int u32x4;

union F16Frag { v16h v; unsigned int u[8]; };

// A-fragment (16x32, f16), ISA 7.12.2 layout:
//   lane 0-15 : M=lane,  K = {0..7, 16..23}   (VGPR v: pair 2v / 16+2v)
//   lane16-31 : M=lane-16, K = {8..15, 24..31}
__device__ __forceinline__ v16h frag_a(const _Float16* p, int ld,
                                       int row0, int k0, int lane) {
  const int hi = (lane >> 4) & 1;
  const _Float16* rp = p + (size_t)(row0 + (lane & 15)) * ld + k0 + 8 * hi;
  F16Frag f;
#pragma unroll
  for (int v = 0; v < 4; ++v)
    f.u[v] = *(const unsigned int*)(rp + 2 * v);
#pragma unroll
  for (int v = 0; v < 4; ++v)
    f.u[4 + v] = *(const unsigned int*)(rp + 16 + 2 * v);
  return f.v;
}

// B-fragment (32x16, f16) built from a row-major [N][K] tile (i.e. B = tile^T):
//   lane 0-15 : N=lane,  K = {0..7, 16..23} ; lane 16-31 : N=lane-16, K = {8..15, 24..31}
__device__ __forceinline__ v16h frag_b(const _Float16* p, int ld,
                                       int n0, int k0, int lane) {
  const int hi = (lane >> 4) & 1;
  const _Float16* rp = p + (size_t)(n0 + (lane & 15)) * ld + k0 + 8 * hi;
  F16Frag f;
#pragma unroll
  for (int v = 0; v < 4; ++v)
    f.u[v] = *(const unsigned int*)(rp + 2 * v);
#pragma unroll
  for (int v = 0; v < 4; ++v)
    f.u[4 + v] = *(const unsigned int*)(rp + 16 + 2 * v);
  return f.v;
}

#define WMMA_F16(a, b, c) \
  __builtin_amdgcn_wmma_f32_16x16x32_f16(false, (a), false, (b), (short)0, (c), false, false)

// Async global->LDS 128-bit copy (CDNA5, ASYNCcnt-tracked, bypasses VGPRs).
// lds_off: LDS byte offset (generic shared addr truncated to 32 bits per the
// LDS aperture rule addr[31:0] == LDS offset). gptr: 64-bit global address.
__device__ __forceinline__ void async_copy_b128(void* lds_ptr, const void* gptr) {
  unsigned lds_off = (unsigned)(size_t)lds_ptr;
  asm volatile("global_load_async_to_lds_b128 %0, %1, off"
               :: "v"(lds_off), "v"(gptr)
               : "memory");
}
__device__ __forceinline__ void wait_async0() {
  asm volatile("s_wait_asynccnt 0" ::: "memory");
}

// ---------------------------------------------------------------------------
// GEMM: C_f16[M,N] = scale*A_f32[M,K] @ W_f32[N,K]^T   (torch Linear)
//   a_mode 0: A row-major [M][K]
//   a_mode 1: A is image (B,512,64,64): A(m,k) = img[b*512*4096 + k*4096 + pix]
//   out_mode 0: C row-major [M][N]
//   out_mode 1: per-batch transposed (256 rows/batch): C[(b*N+n)*256 + m%256]
// Block: 256 threads (8 waves), tile 128(M) x 64(N), K-chunks of 32.
// ---------------------------------------------------------------------------
#define LDA 34  // padded f16 row stride: breaks 4-way LDS bank conflicts

__global__ __launch_bounds__(256) void gemm_qkv(
    const float* __restrict__ A, const float* __restrict__ W,
    _Float16* __restrict__ C, int M, int N, int Kdim,
    int a_mode, int out_mode, float scale) {
  __shared__ _Float16 sA[128 * LDA];
  __shared__ _Float16 sW[64 * LDA];

  const int tid  = threadIdx.x;
  const int lane = tid & 31;
  const int m0   = (tid >> 5) * 16;           // wave's M offset in tile
  const int tileM = blockIdx.x * 128;
  const int tileN = blockIdx.y * 64;

  v8f acc[4] = {};
  for (int k0 = 0; k0 < Kdim; k0 += 32) {
    if (a_mode == 0) {
      for (int i = tid; i < 128 * 32; i += 256) {
        int r = i >> 5, kk = i & 31;
        sA[r * LDA + kk] =
            (_Float16)(A[(size_t)(tileM + r) * Kdim + k0 + kk] * scale);
      }
      // prefetch next K-chunk of A (speculative, GL2 temporal)
      if (k0 + 32 < Kdim)
        __builtin_prefetch(&A[(size_t)(tileM + (tid >> 1)) * Kdim + k0 + 32], 0, 1);
    } else {
      for (int i = tid; i < 128 * 32; i += 256) {
        int r = i & 127, kk = i >> 7;          // consecutive tid -> contig pix
        int m = tileM + r, b = m >> 12, pix = m & 4095;
        sA[r * LDA + kk] =
            (_Float16)(A[((size_t)b * 512 + (k0 + kk)) * 4096 + pix] * scale);
      }
      if (k0 + 32 < Kdim) {
        int m = tileM + (tid & 127), b = m >> 12, pix = m & 4095;
        __builtin_prefetch(&A[((size_t)b * 512 + (k0 + 32 + (tid >> 7))) * 4096 + pix], 0, 1);
      }
    }
    for (int i = tid; i < 64 * 32; i += 256) {
      int r = i >> 5, kk = i & 31;
      sW[r * LDA + kk] = (_Float16)W[(size_t)(tileN + r) * Kdim + k0 + kk];
    }
    __syncthreads();

    v16h a = frag_a(sA, LDA, m0, 0, lane);
#pragma unroll
    for (int nt = 0; nt < 4; ++nt) {
      v16h b = frag_b(sW, LDA, nt * 16, 0, lane);
      acc[nt] = WMMA_F16(a, b, acc[nt]);
    }
    __syncthreads();
  }

  const int hi = (lane >> 4) & 1, col0 = lane & 15;
#pragma unroll
  for (int nt = 0; nt < 4; ++nt) {
#pragma unroll
    for (int i = 0; i < 8; ++i) {
      int gm = tileM + m0 + i + 8 * hi;
      int gn = tileN + nt * 16 + col0;
      _Float16 v = (_Float16)acc[nt][i];
      if (out_mode == 0)
        C[(size_t)gm * N + gn] = v;
      else
        C[((size_t)(gm >> 8) * N + gn) * 256 + (gm & 255)] = v;
    }
  }
}

// ---------------------------------------------------------------------------
// Fused attention: one wave handles a 16-row query tile of one (b, head).
//   scores = (Q/sqrt(dh)) @ K^T  (scale folded into Q) ; softmax(+mask) ; P@V
// LDS: K/V slice 32KB (reused), scores 16KB f32, P 8KB f16  -> 56KB/block.
// K and V^T head-slices staged with async global->LDS (ASYNCcnt); each
// stage is drained (s_wait_asynccnt 0 + barrier) before the phase that reads it.
// ---------------------------------------------------------------------------
__global__ __launch_bounds__(32) void attn_kernel(
    const _Float16* __restrict__ Q,   // [B*4096][512]
    const _Float16* __restrict__ Kg,  // [B*256][512]
    const _Float16* __restrict__ Vt,  // [B][512][256] (per-head transposed V)
    const float* __restrict__ mask,   // [B][4096][256]
    _Float16* __restrict__ O) {       // [B*4096][512]
  __shared__ __align__(16) _Float16 sKV[256 * 64];  // K phase then V phase
  __shared__ float sS[16 * 256];
  __shared__ _Float16 sP[16 * 256];

  const int lane = threadIdx.x;
  int bid = blockIdx.x;
  const int qt = bid & 255;  bid >>= 8;
  const int h  = bid & 7;    bid >>= 3;
  const int b  = bid;
  const int q0 = qt * 16;

  // phase 1: async-stage K head-slice [256 keys][64 dh] (128-bit transfers)
  {
    const _Float16* kb = Kg + (size_t)b * 256 * 512 + h * 64;
    for (int i = lane; i < 2048; i += 32) {
      int row = i >> 3, seg = i & 7;
      async_copy_b128(&sKV[row * 64 + seg * 8], kb + (size_t)row * 512 + seg * 8);
    }
    wait_async0();
  }
  __syncthreads();

  // phase 2: scores = Q @ K^T   (16 x 256)
  const _Float16* qb = Q + (size_t)(b * 4096 + q0) * 512 + h * 64;
  const int hi = (lane >> 4) & 1, col0 = lane & 15;
  for (int nt = 0; nt < 16; ++nt) {
    v8f acc = {};
#pragma unroll
    for (int kc = 0; kc < 64; kc += 32) {
      v16h a  = frag_a(qb, 512, 0, kc, lane);
      v16h bf = frag_b(sKV, 64, nt * 16, kc, lane);
      acc = WMMA_F16(a, bf, acc);
    }
    const int col = nt * 16 + col0;
#pragma unroll
    for (int i = 0; i < 8; ++i) sS[(i + 8 * hi) * 256 + col] = acc[i];
  }
  __syncthreads();

  // phase 3: softmax over 256 keys per row (wave-wide shuffle reduction)
  const float* mrow = mask + ((size_t)b * 4096 + q0) * 256;
  for (int r = 0; r < 16; ++r) {
    float vals[8];
    float mx = -1e30f;
#pragma unroll
    for (int j = 0; j < 8; ++j) {
      int c = lane * 8 + j;
      vals[j] = sS[r * 256 + c] + mrow[(size_t)r * 256 + c];
      mx = fmaxf(mx, vals[j]);
    }
#pragma unroll
    for (int off = 16; off >= 1; off >>= 1)
      mx = fmaxf(mx, __shfl_xor(mx, off, 32));
    float sum = 0.f;
#pragma unroll
    for (int j = 0; j < 8; ++j) { vals[j] = __expf(vals[j] - mx); sum += vals[j]; }
#pragma unroll
    for (int off = 16; off >= 1; off >>= 1) sum += __shfl_xor(sum, off, 32);
    const float inv = 1.0f / sum;
#pragma unroll
    for (int j = 0; j < 8; ++j)
      sP[r * 256 + lane * 8 + j] = (_Float16)(vals[j] * inv);
  }
  __syncthreads();

  // phase 4: async-stage V^T head-slice [64 dh][256 keys] (reuses sKV)
  {
    const _Float16* vb = Vt + ((size_t)b * 512 + h * 64) * 256;
    for (int i = lane; i < 2048; i += 32) {
      int row = i >> 5, seg = i & 31;
      async_copy_b128(&sKV[row * 256 + seg * 8], vb + (size_t)row * 256 + seg * 8);
    }
    wait_async0();
  }
  __syncthreads();

  // phase 5: O = P @ V   (16 x 64)
  _Float16* ob = O + (size_t)(b * 4096 + q0) * 512 + h * 64;
  for (int dt = 0; dt < 4; ++dt) {
    v8f acc = {};
#pragma unroll
    for (int kc = 0; kc < 256; kc += 32) {
      v16h a  = frag_a(sP, 256, 0, kc, lane);
      v16h bf = frag_b(sKV, 256, dt * 16, kc, lane);
      acc = WMMA_F16(a, bf, acc);
    }
    const int col = dt * 16 + col0;
#pragma unroll
    for (int i = 0; i < 8; ++i)
      ob[(size_t)(i + 8 * hi) * 512 + col] = (_Float16)acc[i];
  }
}

// ---------------------------------------------------------------------------
// Output projection: out[b, n, pix] = sum_c O[b*4096+pix][c]*Wo[n][c] + bo[n]
// A already f16; accumulate f32; scattered store into (B,512,64,64).
// ---------------------------------------------------------------------------
__global__ __launch_bounds__(256) void gemm_out(
    const _Float16* __restrict__ A, const float* __restrict__ W,
    const float* __restrict__ bias, float* __restrict__ Out,
    int N, int Kdim) {
  __shared__ _Float16 sA[128 * LDA];
  __shared__ _Float16 sW[64 * LDA];

  const int tid  = threadIdx.x;
  const int lane = tid & 31;
  const int m0   = (tid >> 5) * 16;
  const int tileM = blockIdx.x * 128;
  const int tileN = blockIdx.y * 64;

  v8f acc[4] = {};
  for (int k0 = 0; k0 < Kdim; k0 += 32) {
    for (int i = tid; i < 2048; i += 256) {   // 128 rows x 16 dwords
      int r = i >> 4, kk2 = i & 15;
      ((unsigned int*)sA)[r * (LDA / 2) + kk2] =
          *(const unsigned int*)(A + (size_t)(tileM + r) * Kdim + k0 + kk2 * 2);
    }
    if (k0 + 32 < Kdim)
      __builtin_prefetch(A + (size_t)(tileM + (tid >> 1)) * Kdim + k0 + 32, 0, 1);
    for (int i = tid; i < 64 * 32; i += 256) {
      int r = i >> 5, kk = i & 31;
      sW[r * LDA + kk] = (_Float16)W[(size_t)(tileN + r) * Kdim + k0 + kk];
    }
    __syncthreads();

    v16h a = frag_a(sA, LDA, m0, 0, lane);
#pragma unroll
    for (int nt = 0; nt < 4; ++nt) {
      v16h b = frag_b(sW, LDA, nt * 16, 0, lane);
      acc[nt] = WMMA_F16(a, b, acc[nt]);
    }
    __syncthreads();
  }

  const int hi = (lane >> 4) & 1, col0 = lane & 15;
#pragma unroll
  for (int nt = 0; nt < 4; ++nt) {
    int gn = tileN + nt * 16 + col0;
    float bv = bias[gn];
#pragma unroll
    for (int i = 0; i < 8; ++i) {
      int gm = tileM + m0 + i + 8 * hi;
      int bb = gm >> 12, pix = gm & 4095;
      Out[((size_t)bb * N + gn) * 4096 + pix] = acc[nt][i] + bv;
    }
  }
}

// ---------------------------------------------------------------------------
extern "C" void kernel_launch(void* const* d_in, const int* in_sizes, int n_in,
                              void* d_out, int out_size, void* d_ws, size_t ws_size,
                              hipStream_t stream) {
  const float* image = (const float*)d_in[0];  // (8, 512, 64, 64)
  const float* cond  = (const float*)d_in[1];  // (8, 256, 768)
  const float* mask  = (const float*)d_in[2];  // (8, 4096, 256)
  const float* Wq    = (const float*)d_in[3];  // (512, 512)
  const float* Wk    = (const float*)d_in[4];  // (512, 768)
  const float* Wv    = (const float*)d_in[5];  // (512, 768)
  const float* Wo    = (const float*)d_in[6];  // (512, 512)
  const float* bo    = (const float*)d_in[7];  // (512,)
  float* out = (float*)d_out;                  // (8, 512, 64, 64)

  // workspace carve-up (all f16)
  char* ws = (char*)d_ws;
  _Float16* Qbuf = (_Float16*)ws;                     // 8*4096*512 = 33.5MB
  _Float16* Kbuf = (_Float16*)(ws + (size_t)33554432);          // 2MB
  _Float16* Vtb  = (_Float16*)(ws + (size_t)33554432 + 2097152);// 2MB
  _Float16* Obuf = (_Float16*)(ws + (size_t)33554432 + 2 * 2097152); // 33.5MB

  const float qscale = 0.125f;  // 1/sqrt(dh), dh=64

  // K = cond @ Wk^T              (M=2048, K=768) -> row-major f16
  gemm_qkv<<<dim3(16, 8), 256, 0, stream>>>(cond, Wk, Kbuf, 2048, 512, 768,
                                            0, 0, 1.0f);
  // V = cond @ Wv^T  -> per-head transposed [b][512][256]
  gemm_qkv<<<dim3(16, 8), 256, 0, stream>>>(cond, Wv, Vtb, 2048, 512, 768,
                                            0, 1, 1.0f);
  // Q = (x @ Wq^T) * qscale      (M=32768, K=512), image addressing
  gemm_qkv<<<dim3(256, 8), 256, 0, stream>>>(image, Wq, Qbuf, 32768, 512, 512,
                                             1, 0, qscale);
  // fused attention: B * heads * (4096/16) = 16384 blocks of 1 wave
  attn_kernel<<<16384, 32, 0, stream>>>(Qbuf, Kbuf, Vtb, mask, Obuf);
  // out = Wo @ o + bo, scattered into (B, 512, 64, 64)
  gemm_out<<<dim3(256, 8), 256, 0, stream>>>(Obuf, Wo, bo, out, 512, 512);
}
